// Trifusion_83562883711810
// MI455X (gfx1250) — compile-verified, matrix operand order checked by reading, not run on hardware
//
#include <hip/hip_runtime.h>
#include <hip/hip_bf16.h>
#include <math.h>

// ---------------------------------------------------------------------------
// Types for CDNA5 WMMA (wave32, 16x16x32 bf16 -> f32)
// ---------------------------------------------------------------------------
typedef __attribute__((ext_vector_type(16))) __bf16 v16bf;
typedef __attribute__((ext_vector_type(4)))  __bf16 v4bf;
typedef __attribute__((ext_vector_type(8)))  float  v8f;

#define BM 128
#define BN 128
#define BK 32
#define NT 256

#define NUM_RNA 4000
#define NUM_DIS 2000
#define HIDDEN  256
#define NNODES  6000
#define E_M     2000
#define E_D     1000
#define NPAIRS  16384

// K-permutation inside an LDS row: [K0-7 | K16-23 | K8-15 | K24-31].
// With this layout, lanes 0-15 read permuted elements [0,16) and lanes 16-31
// read [16,32) as ONE contiguous 32B load that is already in WMMA fragment
// element order (elems 0-7 = K kb..kb+7, elems 8-15 = K 16+kb..16+kb+7).
__device__ __forceinline__ int kperm(int k)
{
    return (k & 7) | ((k & 8) << 1) | ((k & 16) >> 1);
}

// ---------------------------------------------------------------------------
// Tile staging: global f32 -> LDS bf16 ([m][perm(k)] for A, [n][perm(k)] for B)
// Fast path (no guards) for fully interior blocks + full K tiles.
// ---------------------------------------------------------------------------
__device__ __forceinline__ void stage_tile(
    const float* __restrict__ A, const float* __restrict__ B,
    int M, int N, int K, int lda, int ldb, int transA,
    int bm, int bn, int kk, int tid, bool mn_full,
    __bf16 (*sA)[BK], __bf16 (*sB)[BK])
{
    const bool fullk = (kk + BK <= K);
    if (mn_full && fullk) {
        if (!transA) {
#pragma unroll
            for (int r = 0; r < 4; ++r) {
                int idx = tid + r * NT;          // [0,1024)
                int m = idx >> 3, k4 = (idx & 7) << 2;
                float4 v = *(const float4*)(A + (size_t)(bm + m) * lda + kk + k4);
                v4bf p; p[0] = (__bf16)v.x; p[1] = (__bf16)v.y;
                        p[2] = (__bf16)v.z; p[3] = (__bf16)v.w;
                *(v4bf*)&sA[m][kperm(k4)] = p;   // one ds_store_b64
            }
        } else {
#pragma unroll
            for (int r = 0; r < 4; ++r) {
                int idx = tid + r * NT;
                int k = idx >> 5, m4 = (idx & 31) << 2;
                int pk = kperm(k);
                float4 v = *(const float4*)(A + (size_t)(kk + k) * lda + bm + m4);
                sA[m4+0][pk] = (__bf16)v.x; sA[m4+1][pk] = (__bf16)v.y;
                sA[m4+2][pk] = (__bf16)v.z; sA[m4+3][pk] = (__bf16)v.w;
            }
        }
#pragma unroll
        for (int r = 0; r < 4; ++r) {
            int idx = tid + r * NT;
            int k = idx >> 5, n4 = (idx & 31) << 2;
            int pk = kperm(k);
            float4 v = *(const float4*)(B + (size_t)(kk + k) * ldb + bn + n4);
            sB[n4+0][pk] = (__bf16)v.x; sB[n4+1][pk] = (__bf16)v.y;
            sB[n4+2][pk] = (__bf16)v.z; sB[n4+3][pk] = (__bf16)v.w;
        }
        return;
    }

    // ---- guarded path (edge blocks / K tail) ----
    if (!transA) {
#pragma unroll
        for (int r = 0; r < 4; ++r) {
            int idx = tid + r * NT;
            int m = idx >> 3, k4 = (idx & 7) << 2;
            int gm = bm + m, gk = kk + k4;
            float x[4] = {0.f, 0.f, 0.f, 0.f};
            if (gm < M) {
                const float* p = A + (size_t)gm * lda + gk;
                if (gk + 4 <= K) { float4 v = *(const float4*)p; x[0]=v.x; x[1]=v.y; x[2]=v.z; x[3]=v.w; }
                else {
#pragma unroll
                    for (int j = 0; j < 4; ++j) if (gk + j < K) x[j] = p[j];
                }
            }
            v4bf pkv; pkv[0] = (__bf16)x[0]; pkv[1] = (__bf16)x[1];
                      pkv[2] = (__bf16)x[2]; pkv[3] = (__bf16)x[3];
            *(v4bf*)&sA[m][kperm(k4)] = pkv;
        }
    } else {
#pragma unroll
        for (int r = 0; r < 4; ++r) {
            int idx = tid + r * NT;
            int k = idx >> 5, m4 = (idx & 31) << 2;
            int gk = kk + k, pk = kperm(k);
            float x[4] = {0.f, 0.f, 0.f, 0.f};
            if (gk < K) {
                const float* p = A + (size_t)gk * lda + bm + m4;
                if (bm + m4 + 4 <= M) { float4 v = *(const float4*)p; x[0]=v.x; x[1]=v.y; x[2]=v.z; x[3]=v.w; }
                else {
#pragma unroll
                    for (int j = 0; j < 4; ++j) if (bm + m4 + j < M) x[j] = p[j];
                }
            }
#pragma unroll
            for (int j = 0; j < 4; ++j) sA[m4+j][pk] = (__bf16)x[j];
        }
    }
#pragma unroll
    for (int r = 0; r < 4; ++r) {
        int idx = tid + r * NT;
        int k = idx >> 5, n4 = (idx & 31) << 2;
        int gk = kk + k, gn = bn + n4, pk = kperm(k);
        float x[4] = {0.f, 0.f, 0.f, 0.f};
        if (gk < K) {
            const float* p = B + (size_t)gk * ldb + gn;
            if (gn + 4 <= N) { float4 v = *(const float4*)p; x[0]=v.x; x[1]=v.y; x[2]=v.z; x[3]=v.w; }
            else {
#pragma unroll
                for (int j = 0; j < 4; ++j) if (gn + j < N) x[j] = p[j];
            }
        }
#pragma unroll
        for (int j = 0; j < 4; ++j) sB[n4+j][pk] = (__bf16)x[j];
    }
}

// ---------------------------------------------------------------------------
// General tiled GEMM: C = op(A) @ B, double-buffered LDS, WMMA compute.
// Epilogue fuses: per-row scale, per-col bias, accumulate-into-C, relu/tanh.
//   non-trans: A[m*lda + k] ; trans: A[k*lda + m]
//   B[k*ldb + n], C[m*ldc + n]
// ---------------------------------------------------------------------------
__global__ __launch_bounds__(NT)
void gemm_wmma_bf16(const float* __restrict__ A, const float* __restrict__ B,
                    float* __restrict__ C, int M, int N, int K,
                    int lda, int ldb, int ldc, int transA,
                    const float* __restrict__ rowscale,
                    const float* __restrict__ bias,
                    int accflag, int act)
{
    __shared__ __bf16 sA[2][BM][BK];   // [buf][m][perm(k)]
    __shared__ __bf16 sB[2][BN][BK];   // [buf][n][perm(k)]

    const int tid  = threadIdx.x;
    const int bm   = blockIdx.y * BM;
    const int bn   = blockIdx.x * BN;
    const int wave = tid >> 5;
    const int lane = tid & 31;
    const int wm   = (wave >> 2) * 64;   // 2 wave-rows (64 rows each)
    const int wn   = (wave & 3)  * 32;   // 4 wave-cols (32 cols each)
    const int r16  = lane & 15;
    const int koff = (lane >> 4) << 4;   // permuted element offset: 0 or 16

    const bool mn_full = (bm + BM <= M) && (bn + BN <= N);
    const int  nk      = (K + BK - 1) / BK;

    v8f acc[4][2];
#pragma unroll
    for (int i = 0; i < 4; ++i)
#pragma unroll
        for (int j = 0; j < 2; ++j)
#pragma unroll
            for (int e = 0; e < 8; ++e) acc[i][j][e] = 0.0f;

    // prologue: stage tile 0
    stage_tile(A, B, M, N, K, lda, ldb, transA, bm, bn, 0, tid, mn_full,
               sA[0], sB[0]);

    int cur = 0;
    for (int t = 0; t < nk; ++t) {
        __syncthreads();

        // 1) fragment loads: each fragment is ONE contiguous 32B LDS read
        //    already in WMMA element order (thanks to the K permutation)
        v16bf af[4], bfr[2];
#pragma unroll
        for (int mt = 0; mt < 4; ++mt)
            af[mt] = *(const v16bf*)&sA[cur][wm + mt * 16 + r16][koff];
#pragma unroll
        for (int nt = 0; nt < 2; ++nt)
            bfr[nt] = *(const v16bf*)&sB[cur][wn + nt * 16 + r16][koff];

        // 2) stage tile t+1 into the other buffer (global loads overlap WMMA)
        if (t + 1 < nk) {
            int kn = (t + 1) * BK;
            stage_tile(A, B, M, N, K, lda, ldb, transA, bm, bn, kn, tid, mn_full,
                       sA[cur ^ 1], sB[cur ^ 1]);
            if (t + 2 < nk) {   // pull tile t+2 toward GL2
                int kp = (t + 2) * BK;
                __builtin_prefetch(A + (transA ? ((size_t)(kp + (tid >> 5)) * lda + bm + lane * 4)
                                               : ((size_t)(bm + (tid >> 1)) * lda + kp)), 0, 1);
                __builtin_prefetch(B + (size_t)(kp + (tid >> 5)) * ldb + bn + lane * 4, 0, 1);
            }
        }

        // 3) WMMA chain
#pragma unroll
        for (int mt = 0; mt < 4; ++mt)
#pragma unroll
            for (int nt = 0; nt < 2; ++nt)
                acc[mt][nt] = __builtin_amdgcn_wmma_f32_16x16x32_bf16(
                    false, af[mt], false, bfr[nt], (short)0, acc[mt][nt], false, false);

        cur ^= 1;
    }

    // ---------------- fused epilogue ----------------
    const int rowoff = (lane >> 4) << 3;   // lanes 0-15 -> M+0..7 ; 16-31 -> M+8..15
#pragma unroll
    for (int mt = 0; mt < 4; ++mt) {
#pragma unroll
        for (int nt = 0; nt < 2; ++nt) {
            int gcol = bn + wn + nt * 16 + r16;
            if (gcol >= N) continue;
            float bv = bias ? bias[gcol] : 0.0f;
#pragma unroll
            for (int v = 0; v < 8; ++v) {
                int grow = bm + wm + mt * 16 + rowoff + v;
                if (grow >= M) continue;
                float val = acc[mt][nt][v];
                if (rowscale) val *= rowscale[grow];
                val += bv;
                if (accflag)  val += C[(size_t)grow * ldc + gcol];
                if (act == 1)      val = fmaxf(val, 0.0f);
                else if (act == 2) val = tanhf(val);
                C[(size_t)grow * ldc + gcol] = val;
            }
        }
    }
}

// ---------------------------------------------------------------------------
// Helpers: degree reductions, scaling, gather, final dot
// ---------------------------------------------------------------------------
// block-per-row sum; mode 0: safe_inv ; mode 1: clip(x,1)^-0.5
__global__ __launch_bounds__(256)
void rowsum_inv(const float* __restrict__ src, int ncols, float* __restrict__ out, int mode)
{
    __shared__ float red[256];
    int row = blockIdx.x;
    const float* p = src + (size_t)row * ncols;
    float s = 0.f;
    for (int c = threadIdx.x; c < ncols; c += 256) s += p[c];
    red[threadIdx.x] = s;
    __syncthreads();
    for (int off = 128; off > 0; off >>= 1) {
        if (threadIdx.x < off) red[threadIdx.x] += red[threadIdx.x + off];
        __syncthreads();
    }
    if (threadIdx.x == 0) {
        float x = red[0];
        out[row] = (mode == 0) ? ((x > 0.f) ? 1.f / fmaxf(x, 1e-12f) : 0.f)
                               : rsqrtf(fmaxf(x, 1.f));
    }
}

// thread-per-column sum (coalesced across threads), safe_inv
__global__ __launch_bounds__(256)
void colsum_inv(const float* __restrict__ src, int nrows, int ncols, float* __restrict__ out)
{
    int c = blockIdx.x * blockDim.x + threadIdx.x;
    if (c >= ncols) return;
    float s = 0.f;
    for (int r = 0; r < nrows; ++r) s += src[(size_t)r * ncols + c];
    out[c] = (s > 0.f) ? 1.f / fmaxf(s, 1e-12f) : 0.f;
}

__global__ __launch_bounds__(256)
void scale_rows(const float* __restrict__ src, const float* __restrict__ vec,
                float* __restrict__ dst, int ncols, long long total)
{
    long long i = (long long)blockIdx.x * blockDim.x + threadIdx.x;
    if (i >= total) return;
    int r = (int)(i / ncols);
    dst[i] = src[i] * vec[r];
}

__global__ __launch_bounds__(256)
void relu_copy(const float* __restrict__ src, float* __restrict__ dst, long long total)
{
    long long i = (long long)blockIdx.x * blockDim.x + threadIdx.x;
    if (i >= total) return;
    dst[i] = fmaxf(src[i], 0.f);
}

// z[p,:] = emb[rna_idx[p],:] * emb[NUM_RNA + dis_idx[p],:]
__global__ __launch_bounds__(HIDDEN)
void had_gather(const float* __restrict__ emb, const int* __restrict__ ridx,
                const int* __restrict__ didx, float* __restrict__ z)
{
    int p = blockIdx.x, c = threadIdx.x;
    float a = emb[(size_t)ridx[p] * HIDDEN + c];
    float b = emb[(size_t)(NUM_RNA + didx[p]) * HIDDEN + c];
    z[(size_t)p * HIDDEN + c] = a * b;
}

// out[p] = z3[p,:] . w4 + c4
__global__ __launch_bounds__(256)
void final_dot(const float* __restrict__ z3, const float* __restrict__ w4,
               const float* __restrict__ c4, float* __restrict__ out, int npairs)
{
    int p = blockIdx.x * blockDim.x + threadIdx.x;
    if (p >= npairs) return;
    const float* zr = z3 + (size_t)p * 32;
    float s = c4[0];
#pragma unroll
    for (int k = 0; k < 32; ++k) s += zr[k] * w4[k];
    out[p] = s;
}

// ---------------------------------------------------------------------------
static inline void gemm(hipStream_t s, const float* A, const float* B, float* C,
                        int M, int N, int K, int lda, int ldb, int ldc, int transA,
                        const float* rowscale, const float* bias, int accf, int act)
{
    dim3 grid((N + BN - 1) / BN, (M + BM - 1) / BM);
    gemm_wmma_bf16<<<grid, NT, 0, s>>>(A, B, C, M, N, K, lda, ldb, ldc, transA,
                                       rowscale, bias, accf, act);
}

extern "C" void kernel_launch(void* const* d_in, const int* in_sizes, int n_in,
                              void* d_out, int out_size, void* d_ws, size_t ws_size,
                              hipStream_t stream)
{
    (void)in_sizes; (void)n_in; (void)out_size; (void)ws_size;

    const float* Xm     = (const float*)d_in[0];
    const float* Xd     = (const float*)d_in[1];
    const float* trip_m = (const float*)d_in[2];   // [E_M, NUM_RNA]
    const float* trip_d = (const float*)d_in[3];   // [E_D, NUM_DIS]
    const float* adj    = (const float*)d_in[4];   // [N, N] symmetric
    const int*   ridx   = (const int*)d_in[5];
    const int*   didx   = (const int*)d_in[6];
    const float* th1m = (const float*)d_in[7];  const float* b1m = (const float*)d_in[8];
    const float* th2m = (const float*)d_in[9];  const float* b2m = (const float*)d_in[10];
    const float* th1d = (const float*)d_in[11]; const float* b1d = (const float*)d_in[12];
    const float* th2d = (const float*)d_in[13]; const float* b2d = (const float*)d_in[14];
    const float* gw   = (const float*)d_in[15]; const float* gb  = (const float*)d_in[16];
    const float* w1   = (const float*)d_in[17]; const float* c1  = (const float*)d_in[18];
    const float* w2   = (const float*)d_in[19]; const float* c2  = (const float*)d_in[20];
    const float* w3   = (const float*)d_in[21]; const float* c3  = (const float*)d_in[22];
    const float* w4   = (const float*)d_in[23]; const float* c4  = (const float*)d_in[24];

    // ---------------- workspace layout (floats) ----------------
    float* ws   = (float*)d_ws;
    float* EMB  = ws;                          // [6000,256]  X then final emb
    float* RES  = EMB + (size_t)NNODES * HIDDEN;
    float* FB   = RES + (size_t)NNODES * HIDDEN;   // f / h1 temp
    float* TB   = FB  + (size_t)NNODES * HIDDEN;   // f*norm / Xt temp (>= 4000*256)
    float* EB   = TB  + (size_t)NNODES * HIDDEN;   // hyperedge temp [2000,256]
    float* VEC  = EB  + (size_t)E_M * HIDDEN;
    float* BinvM = VEC;            // 2000
    float* DinvM = VEC + 2000;     // 4000
    float* BinvD = VEC + 6000;     // 1000
    float* DinvD = VEC + 7000;     // 2000
    float* NORM  = VEC + 9000;     // 6000
    float* Z1 = VEC + 16000;                        // [16384,128]
    float* Z2 = Z1 + (size_t)NPAIRS * 128;          // [16384,64]
    float* Z3 = Z2 + (size_t)NPAIRS * 64;           // [16384,32]
    float* Z0 = RES;   // [16384,256] overlaps RES/FB/TB (dead after emb is final)

    // ---------------- degrees / norms ----------------
    rowsum_inv<<<E_M,    256, 0, stream>>>(trip_m, NUM_RNA, BinvM, 0);
    colsum_inv<<<(NUM_RNA + 255) / 256, 256, 0, stream>>>(trip_m, E_M, NUM_RNA, DinvM);
    rowsum_inv<<<E_D,    256, 0, stream>>>(trip_d, NUM_DIS, BinvD, 0);
    colsum_inv<<<(NUM_DIS + 255) / 256, 256, 0, stream>>>(trip_d, E_D, NUM_DIS, DinvD);
    rowsum_inv<<<NNODES, 256, 0, stream>>>(adj, NNODES, NORM, 1);

    // ---------------- RNA hypergraph branch ----------------
    // hconv1: Xt = Xm@th1m ; e = (trip@Xt)*Binv ; h1 = relu((tripT@e)*Dinv + b1m)
    gemm(stream, Xm,     th1m, TB,  NUM_RNA, HIDDEN, NUM_RNA, NUM_RNA, HIDDEN, HIDDEN, 0, nullptr, nullptr, 0, 0);
    gemm(stream, trip_m, TB,   EB,  E_M,     HIDDEN, NUM_RNA, NUM_RNA, HIDDEN, HIDDEN, 0, BinvM,   nullptr, 0, 0);
    gemm(stream, trip_m, EB,   FB,  NUM_RNA, HIDDEN, E_M,     NUM_RNA, HIDDEN, HIDDEN, 1, DinvM,   b1m,     0, 1);
    // hconv2 -> EMB rows [0,4000)
    gemm(stream, FB,     th2m, TB,  NUM_RNA, HIDDEN, HIDDEN,  HIDDEN,  HIDDEN, HIDDEN, 0, nullptr, nullptr, 0, 0);
    gemm(stream, trip_m, TB,   EB,  E_M,     HIDDEN, NUM_RNA, NUM_RNA, HIDDEN, HIDDEN, 0, BinvM,   nullptr, 0, 0);
    gemm(stream, trip_m, EB,   EMB, NUM_RNA, HIDDEN, E_M,     NUM_RNA, HIDDEN, HIDDEN, 1, DinvM,   b2m,     0, 0);

    // ---------------- disease hypergraph branch ----------------
    gemm(stream, Xd,     th1d, TB,  NUM_DIS, HIDDEN, NUM_DIS, NUM_DIS, HIDDEN, HIDDEN, 0, nullptr, nullptr, 0, 0);
    gemm(stream, trip_d, TB,   EB,  E_D,     HIDDEN, NUM_DIS, NUM_DIS, HIDDEN, HIDDEN, 0, BinvD,   nullptr, 0, 0);
    gemm(stream, trip_d, EB,   FB,  NUM_DIS, HIDDEN, E_D,     NUM_DIS, HIDDEN, HIDDEN, 1, DinvD,   b1d,     0, 1);
    gemm(stream, FB,     th2d, TB,  NUM_DIS, HIDDEN, HIDDEN,  HIDDEN,  HIDDEN, HIDDEN, 0, nullptr, nullptr, 0, 0);
    gemm(stream, trip_d, TB,   EB,  E_D,     HIDDEN, NUM_DIS, NUM_DIS, HIDDEN, HIDDEN, 0, BinvD,   nullptr, 0, 0);
    gemm(stream, trip_d, EB,   EMB + (size_t)NUM_RNA * HIDDEN,
                              NUM_DIS, HIDDEN, E_D,     NUM_DIS, HIDDEN, HIDDEN, 1, DinvD,   b2d,     0, 0);

    // ---------------- Parallel GCN on association graph ----------------
    const long long NE = (long long)NNODES * HIDDEN;
    gemm(stream, EMB, gw, RES, NNODES, HIDDEN, HIDDEN, HIDDEN, HIDDEN, HIDDEN, 0, nullptr, gb, 0, 0);
    const float* f = EMB;
    for (int hop = 0; hop < 3; ++hop) {
        scale_rows<<<(int)((NE + 255) / 256), 256, 0, stream>>>(f, NORM, TB, HIDDEN, NE);
        gemm(stream, adj, TB, FB, NNODES, HIDDEN, NNODES, NNODES, HIDDEN, HIDDEN, 0, NORM, nullptr, 0, 0);
        gemm(stream, FB,  gw, RES, NNODES, HIDDEN, HIDDEN, HIDDEN, HIDDEN, HIDDEN, 0, nullptr, gb, 1, 0);
        f = FB;
    }
    relu_copy<<<(int)((NE + 255) / 256), 256, 0, stream>>>(RES, EMB, NE);

    // ---------------- pairwise decoder ----------------
    had_gather<<<NPAIRS, HIDDEN, 0, stream>>>(EMB, ridx, didx, Z0);
    gemm(stream, Z0, w1, Z1, NPAIRS, 128, 256, 256, 128, 128, 0, nullptr, c1, 0, 2);
    gemm(stream, Z1, w2, Z2, NPAIRS, 64,  128, 128, 64,  64,  0, nullptr, c2, 0, 2);
    gemm(stream, Z2, w3, Z3, NPAIRS, 32,  64,  64,  32,  32,  0, nullptr, c3, 0, 2);
    final_dot<<<(NPAIRS + 255) / 256, 256, 0, stream>>>(Z3, w4, c4, (float*)d_out, NPAIRS);
}
